// EncoderStack_7722351199136
// MI455X (gfx1250) — matrix operand order repaired and checked by reading, not compiled, not run on hardware
//
#include <hip/hip_runtime.h>
#include <hip/hip_bf16.h>
#include <stdint.h>

// ---------------------------------------------------------------------------
// Types for CDNA5 WMMA (wave32): bf16 16x16x32, f32 accumulate.
// ---------------------------------------------------------------------------
typedef __bf16 bf16;
typedef __attribute__((ext_vector_type(16))) __bf16 v16bf;
typedef __attribute__((ext_vector_type(8)))  float  v8f;
typedef unsigned int u32;
typedef __attribute__((ext_vector_type(4))) u32 v4u;
typedef __attribute__((ext_vector_type(8))) int v8i;
typedef __attribute__((ext_vector_type(4))) int v4i;

#define DEV __device__ __forceinline__

DEV v8f v8f_zero() { v8f z = {0.f,0.f,0.f,0.f,0.f,0.f,0.f,0.f}; return z; }

// ---------------------------------------------------------------------------
// CDNA5 async global->LDS copy (ASYNCcnt-tracked), GVS addressing:
//   lds[vdst(lane)] = mem[saddr + vaddr32(lane)]
// ---------------------------------------------------------------------------
DEV void async_copy_b128(u32 ldsByteAddr, const void* base, u32 gByteOff) {
  asm volatile("global_load_async_to_lds_b128 %0, %1, %2"
               :: "v"(ldsByteAddr), "v"(gByteOff), "s"(base)
               : "memory");
}
DEV void async_wait0() { asm volatile("s_wait_asynccnt 0x0" ::: "memory"); }

// ---------------------------------------------------------------------------
// CDNA5 Tensor Data Mover: DMA a 2-D bf16 tile (tileW x tileH elements) from a
// row-major tensor (rowStride elements between rows) into LDS, with the TDM
// padding feature producing a padded LDS pitch. Issued once per wave; tracked
// by TENSORcnt. D# bitfields per CDNA5 ISA.
// This toolchain exposes the 6-arg builtin: (v4u, v8i, v4i, v4i, v8i, i32).
//   padIntervalCode: pad after 2^(c+1) DWORDs ; padAmountCode: pad (c+1) DWORDs
// ---------------------------------------------------------------------------
DEV void tdm_load_2d_bf16(u32 ldsByteAddr, const void* gaddr,
                          u32 tileW, u32 tileH, u32 rowStride, u32 tensorH,
                          u32 padIntervalCode, u32 padAmountCode, bool padEn) {
  unsigned long long ga = (unsigned long long)gaddr;
  v4u g0;
  g0[0] = 1u;                                        // count=1 (valid user D#)
  g0[1] = ldsByteAddr;                               // lds_addr [63:32]
  g0[2] = (u32)(ga & 0xFFFFFFFFu);                   // global_addr [95:64]
  g0[3] = (u32)((ga >> 32) & 0x01FFFFFFu) | (2u << 30);  // addr[56:32] | type=2
  v8i g1;
  g1[0] = (int)((1u << 16)                           // data_size = 2 bytes
                | ((padEn ? 1u : 0u) << 20)          // pad_enable
                | (padIntervalCode << 22)
                | (padAmountCode << 25));
  g1[1] = (int)((rowStride & 0xFFFFu) << 16);        // tensor_dim0[15:0]
  g1[2] = (int)(((rowStride >> 16) & 0xFFFFu)        // tensor_dim0[31:16]
                | ((tensorH & 0xFFFFu) << 16));      // tensor_dim1[15:0]
  g1[3] = (int)(((tensorH >> 16) & 0xFFFFu)          // tensor_dim1[31:16]
                | ((tileW & 0xFFFFu) << 16));        // tile_dim0
  g1[4] = (int)(tileH & 0xFFFFu);                    // tile_dim1 (tile_dim2=0)
  g1[5] = (int)rowStride;                            // tensor_dim0_stride[31:0]
  g1[6] = 0;                                         // stride hi / dim1_stride
  g1[7] = 0;
  v4i z4 = {0, 0, 0, 0};
  v8i z8 = {0, 0, 0, 0, 0, 0, 0, 0};
  __builtin_amdgcn_tensor_load_to_lds(g0, g1, z4, z4, z8, 0);
}
DEV void tdm_wait0() { __builtin_amdgcn_s_wait_tensorcnt(0); }

// A-matrix 16x32 bf16 fragment from row-major LDS tile.
// ISA layout: lane L: M = L&15, half = L>>4; VGPR v holds K = (v>>2)*16 + half*8 + 2*(v&3), +1
DEV v16bf frag_a_lds(const bf16* lds, int row0, int lda, int k0, int lane) {
  int M = lane & 15, half = lane >> 4;
  const bf16* base = lds + (row0 + M) * lda + k0;
  v16bf a;
  uint32_t* au = reinterpret_cast<uint32_t*>(&a);
#pragma unroll
  for (int v = 0; v < 8; ++v) {
    int K = ((v >> 2) << 4) + (half << 3) + ((v & 3) << 1);
    au[v] = *reinterpret_cast<const uint32_t*>(base + K);
  }
  return a;
}

// B-matrix 32x16 bf16 fragment from transposed LDS tile ldsT[n][k].
// ISA layout: lane L: N = L&15, half = L>>4; VGPR v holds K = (v>>2)*16 + half*8 + 2*(v&3), +1
DEV v16bf frag_b_ldsT(const bf16* ldsT, int col0, int ldt, int k0, int lane) {
  int N = lane & 15, half = lane >> 4;
  const bf16* base = ldsT + (col0 + N) * ldt + k0;
  v16bf b;
  uint32_t* bu = reinterpret_cast<uint32_t*>(&b);
#pragma unroll
  for (int v = 0; v < 8; ++v)
    bu[v] = *reinterpret_cast<const uint32_t*>(base + ((v >> 2) << 4) + (half << 3) + ((v & 3) << 1));
  return b;
}

DEV v8f wmma_bf16(v16bf a, v16bf b, v8f c) {
  return __builtin_amdgcn_wmma_f32_16x16x32_bf16(false, a, false, b, (short)0, c,
                                                 false, false);
}

// ---------------------------------------------------------------------------
// f32 -> bf16 conversion
// ---------------------------------------------------------------------------
__global__ __launch_bounds__(256) void f32_to_bf16_kernel(
    const float* __restrict__ src, bf16* __restrict__ dst, long n) {
  for (long i = (long)blockIdx.x * blockDim.x + threadIdx.x; i < n;
       i += (long)gridDim.x * blockDim.x)
    dst[i] = (bf16)src[i];
}

// ---------------------------------------------------------------------------
// Generic batched bf16 GEMM: C[z] = A[z/aDiv] (MxK, row-major) * B[z%bMod] (KxN)
// Block tile 128x64, 256 threads = 8 waves (4x2), each wave 32x32.
// A tile staged via async global->LDS (ASYNCcnt); B tile transposed manually.
// Epilogues: 0 = bf16 store, 1 = f32 store, 2 = bias+relu -> bf16, 3 = bias -> f32
// Requires M%128==0, N%64==0, K%32==0 (true for all uses here).
// ---------------------------------------------------------------------------
template <int EP>
__global__ __launch_bounds__(256) void gemm_bf16_kernel(
    const bf16* __restrict__ A, const bf16* __restrict__ Bm,
    void* __restrict__ Cout, const float* __restrict__ bias,
    int M, int N, int K,
    long aStrideZ, int aDiv, long bStrideZ, int bMod, long cStrideZ) {
  constexpr int LDA = 40;  // 32 + pad; row pitch 80B (16B aligned)
  constexpr int LDB = 34;  // transposed B tile [n][k]; 4B-aligned pairs
  __shared__ bf16 sA[128 * LDA];
  __shared__ bf16 sB[64 * LDB];

  const int tid = threadIdx.x;
  const int lane = tid & 31;
  const int wid = tid >> 5;
  const int wm = wid >> 1;  // 0..3
  const int wn = wid & 1;   // 0..1
  const int z = blockIdx.z;
  const bf16* Ab = A + (long)(z / aDiv) * aStrideZ;
  const bf16* Bb = Bm + (long)(z % bMod) * bStrideZ;
  const int m0 = blockIdx.x * 128;
  const int n0 = blockIdx.y * 64;

  v8f acc[2][2];
#pragma unroll
  for (int i = 0; i < 2; ++i)
#pragma unroll
    for (int j = 0; j < 2; ++j) acc[i][j] = v8f_zero();

  const int ar = tid >> 1, ac = (tid & 1) * 16;   // A stage: 128 rows x 32 cols
  const int bk = tid >> 3, bc = (tid & 7) * 8;    // B stage: 32 rows x 64 cols
  const u32 ldsA0 = (u32)(uintptr_t)&sA[ar * LDA + ac];

  for (int k0 = 0; k0 < K; k0 += 32) {
    __syncthreads();
    {
      const u32 gOff = (u32)(((long)(m0 + ar) * K + k0 + ac) * 2);
      async_copy_b128(ldsA0,      Ab, gOff);
      async_copy_b128(ldsA0 + 16, Ab, gOff + 16);
      if (k0 + 32 < K) __builtin_prefetch(Ab + (long)(m0 + ar) * K + k0 + 32, 0, 0);
    }
    {
      const bf16* g = Bb + (long)(k0 + bk) * N + n0 + bc;
      alignas(16) bf16 tmp[8];
      *reinterpret_cast<uint4*>(tmp) = *reinterpret_cast<const uint4*>(g);
#pragma unroll
      for (int j = 0; j < 8; ++j) sB[(bc + j) * LDB + bk] = tmp[j];
    }
    async_wait0();
    __syncthreads();

    v16bf bfrag[2];
#pragma unroll
    for (int tn = 0; tn < 2; ++tn)
      bfrag[tn] = frag_b_ldsT(sB, wn * 32 + tn * 16, LDB, 0, lane);
#pragma unroll
    for (int tm = 0; tm < 2; ++tm) {
      v16bf afrag = frag_a_lds(sA, wm * 32 + tm * 16, LDA, 0, lane);
#pragma unroll
      for (int tn = 0; tn < 2; ++tn)
        acc[tm][tn] = wmma_bf16(afrag, bfrag[tn], acc[tm][tn]);
    }
  }

  const long cz = (long)z * cStrideZ;
#pragma unroll
  for (int tm = 0; tm < 2; ++tm)
#pragma unroll
    for (int tn = 0; tn < 2; ++tn) {
      const int col = n0 + wn * 32 + tn * 16 + (lane & 15);
      const float bv = (EP >= 2) ? bias[col] : 0.f;
#pragma unroll
      for (int r = 0; r < 8; ++r) {
        const int row = m0 + wm * 32 + tm * 16 + r + ((lane >> 4) << 3);
        const long idx = cz + (long)row * N + col;
        const float vv = acc[tm][tn][r];
        if (EP == 0)      ((bf16*)Cout)[idx]  = (bf16)vv;
        else if (EP == 1) ((float*)Cout)[idx] = vv;
        else if (EP == 2) ((bf16*)Cout)[idx]  = (bf16)fmaxf(vv + bv, 0.f);
        else              ((float*)Cout)[idx] = vv + bv;
      }
    }
}

// ---------------------------------------------------------------------------
// Attention pass 1: per-column softmax stats (softmax is over the QUERY axis).
// Each block owns 64 columns s of one (b,h) head and sweeps all T query rows,
// maintaining running (max, sumexp) per column. stats[(bh*T+s)*2] = {m, Z}.
// K tile staged by the Tensor Data Mover; Q tiles by async global->LDS.
// ---------------------------------------------------------------------------
__global__ __launch_bounds__(256) void attn_stats_kernel(
    const bf16* __restrict__ Q, const bf16* __restrict__ Kmat,
    float* __restrict__ stats, int T) {
  constexpr int LQ = 72;  // 64 + pad (16B-aligned rows)
  constexpr int LK = 72;  // K rows as B^T tile [s][d]; TDM-padded pitch
  __shared__ bf16 sQ[128 * LQ];
  __shared__ bf16 sK[64 * LK];
  __shared__ float m_run[64], z_run[64];
  __shared__ float pm[4][64], pz[4][64];

  const int tid = threadIdx.x, lane = tid & 31, wid = tid >> 5;
  const int wm = wid >> 1, wn = wid & 1;
  const int bh = blockIdx.y;
  const int s0 = blockIdx.x * 64;
  const long headOff = (long)bh * T * 64;
  const float scale = 0.125f;  // 1/sqrt(64)

  if (tid < 64) { m_run[tid] = -1e30f; z_run[tid] = 0.f; }

  // K tile: rows s0..s0+63, DMA'd by TDM into the 72-element padded pitch.
  // pad after 2^(4+1)=32 DWORDs (=128B row), pad amount 3+1=4 DWORDs (=16B).
  if (wid == 0) {
    tdm_load_2d_bf16((u32)(uintptr_t)sK, Kmat + headOff + (long)s0 * 64,
                     /*tileW=*/64, /*tileH=*/64, /*rowStride=*/64,
                     /*tensorH=*/(u32)T, /*padI=*/4, /*padA=*/3, true);
    tdm_wait0();
  }

  const int qr = tid >> 1, qc = (tid & 1) * 32;
  const u32 ldsQ0 = (u32)(uintptr_t)&sQ[qr * LQ + qc];

  for (int q0 = 0; q0 < T; q0 += 128) {
    __syncthreads();
    {
      const u32 gOff = (u32)((headOff + (long)(q0 + qr) * 64 + qc) * 2);
#pragma unroll
      for (int j = 0; j < 4; ++j)
        async_copy_b128(ldsQ0 + 16 * j, Q, gOff + 16 * j);
      async_wait0();
    }
    __syncthreads();

    v8f sacc[2][2];
#pragma unroll
    for (int i = 0; i < 2; ++i)
#pragma unroll
      for (int j = 0; j < 2; ++j) sacc[i][j] = v8f_zero();

#pragma unroll
    for (int k0 = 0; k0 < 64; k0 += 32) {
      v16bf bfrag[2];
#pragma unroll
      for (int tn = 0; tn < 2; ++tn)
        bfrag[tn] = frag_b_ldsT(sK, wn * 32 + tn * 16, LK, k0, lane);
#pragma unroll
      for (int tm = 0; tm < 2; ++tm) {
        v16bf afrag = frag_a_lds(sQ, wm * 32 + tm * 16, LQ, k0, lane);
#pragma unroll
        for (int tn = 0; tn < 2; ++tn)
          sacc[tm][tn] = wmma_bf16(afrag, bfrag[tn], sacc[tm][tn]);
      }
    }

    // Per-lane column stats over the 16 rows this lane holds per column.
#pragma unroll
    for (int tn = 0; tn < 2; ++tn) {
      float mloc = -1e30f;
#pragma unroll
      for (int tm = 0; tm < 2; ++tm)
#pragma unroll
        for (int r = 0; r < 8; ++r) mloc = fmaxf(mloc, sacc[tm][tn][r] * scale);
      float zloc = 0.f;
#pragma unroll
      for (int tm = 0; tm < 2; ++tm)
#pragma unroll
        for (int r = 0; r < 8; ++r) zloc += __expf(sacc[tm][tn][r] * scale - mloc);
      // lanes L and L^16 hold the same column (different row halves)
      float mo = __shfl_xor(mloc, 16, 32);
      float zo = __shfl_xor(zloc, 16, 32);
      float mn = fmaxf(mloc, mo);
      float zn = zloc * __expf(mloc - mn) + zo * __expf(mo - mn);
      if (lane < 16) {
        pm[wm][wn * 32 + tn * 16 + lane] = mn;
        pz[wm][wn * 32 + tn * 16 + lane] = zn;
      }
    }
    __syncthreads();
    if (tid < 64) {
      float m = m_run[tid], zz = z_run[tid];
#pragma unroll
      for (int i = 0; i < 4; ++i) {
        float mi = pm[i][tid], zi = pz[i][tid];
        float mn = fmaxf(m, mi);
        zz = zz * __expf(m - mn) + zi * __expf(mi - mn);
        m = mn;
      }
      m_run[tid] = m; z_run[tid] = zz;
    }
  }
  __syncthreads();
  if (tid < 64) {
    long o = ((long)bh * T + s0 + tid) * 2;
    stats[o] = m_run[tid];
    stats[o + 1] = z_run[tid];
  }
}

// ---------------------------------------------------------------------------
// Attention pass 2: out[q,v] = sum_s exp(S[q,s]-m_s)/Z_s * V[s,v]
// Recompute S tiles, normalize per column s, WMMA against V. Fused, never
// touching HBM with the 2048x2048 score matrix. Stores heads as (b,q,h*dv) bf16.
// Q staged by async global->LDS; K tiles by TDM; V transposed manually.
// ---------------------------------------------------------------------------
__global__ __launch_bounds__(256) void attn_av_kernel(
    const bf16* __restrict__ Q, const bf16* __restrict__ Kmat,
    const bf16* __restrict__ V, const float* __restrict__ stats,
    bf16* __restrict__ heads, int T, int H) {
  constexpr int LQ = 72, LK = 72, LV = 72, LP = 72;
  __shared__ bf16 sQ[128 * LQ];
  __shared__ bf16 sK[64 * LK];
  __shared__ bf16 sVT[64 * LV];  // [v][s]
  __shared__ bf16 sP[128 * LP];  // normalized probs, row-major [q][s]
  __shared__ float mcol[64], izcol[64];

  const int tid = threadIdx.x, lane = tid & 31, wid = tid >> 5;
  const int wm = wid >> 1, wn = wid & 1;
  const int bh = blockIdx.y;
  const int b = bh / H, h = bh % H;
  const int q0 = blockIdx.x * 128;
  const long headOff = (long)bh * T * 64;
  const float scale = 0.125f;

  {  // Q block, loaded once via async copy
    int r = tid >> 1, c = (tid & 1) * 32;
    const u32 ldsQ0 = (u32)(uintptr_t)&sQ[r * LQ + c];
    const u32 gOff = (u32)((headOff + (long)(q0 + r) * 64 + c) * 2);
#pragma unroll
    for (int j = 0; j < 4; ++j)
      async_copy_b128(ldsQ0 + 16 * j, Q, gOff + 16 * j);
    async_wait0();
  }

  v8f oacc[2][2];
#pragma unroll
  for (int i = 0; i < 2; ++i)
#pragma unroll
    for (int j = 0; j < 2; ++j) oacc[i][j] = v8f_zero();

  for (int s0 = 0; s0 < T; s0 += 64) {
    __syncthreads();
    if (wid == 0) {  // K tile via Tensor Data Mover (padded pitch 72)
      tdm_load_2d_bf16((u32)(uintptr_t)sK, Kmat + headOff + (long)s0 * 64,
                       64, 64, 64, (u32)T, 4, 3, true);
    }
    {
      int r = tid >> 2, c0 = (tid & 3) * 16;
      const bf16* gv = V + headOff + (long)(s0 + r) * 64 + c0;
      alignas(16) bf16 tv[16];
      *reinterpret_cast<uint4*>(tv)     = *reinterpret_cast<const uint4*>(gv);
      *reinterpret_cast<uint4*>(tv + 8) = *reinterpret_cast<const uint4*>(gv + 8);
#pragma unroll
      for (int j = 0; j < 16; ++j) sVT[(c0 + j) * LV + r] = tv[j];
    }
    if (tid < 64) {
      long so = ((long)bh * T + s0 + tid) * 2;
      mcol[tid] = stats[so];
      izcol[tid] = 1.f / stats[so + 1];
    }
    if (wid == 0) tdm_wait0();
    __syncthreads();

    v8f sacc[2][2];
#pragma unroll
    for (int i = 0; i < 2; ++i)
#pragma unroll
      for (int j = 0; j < 2; ++j) sacc[i][j] = v8f_zero();

#pragma unroll
    for (int k0 = 0; k0 < 64; k0 += 32) {
      v16bf bfrag[2];
#pragma unroll
      for (int tn = 0; tn < 2; ++tn)
        bfrag[tn] = frag_b_ldsT(sK, wn * 32 + tn * 16, LK, k0, lane);
#pragma unroll
      for (int tm = 0; tm < 2; ++tm) {
        v16bf afrag = frag_a_lds(sQ, wm * 32 + tm * 16, LQ, k0, lane);
#pragma unroll
        for (int tn = 0; tn < 2; ++tn)
          sacc[tm][tn] = wmma_bf16(afrag, bfrag[tn], sacc[tm][tn]);
      }
    }

    // P = exp(S*scale - m_col) * (1/Z_col) -> LDS (bf16, row-major)
#pragma unroll
    for (int tm = 0; tm < 2; ++tm)
#pragma unroll
      for (int tn = 0; tn < 2; ++tn) {
        const int col = wn * 32 + tn * 16 + (lane & 15);
        const float mm = mcol[col], iz = izcol[col];
#pragma unroll
        for (int r = 0; r < 8; ++r) {
          const int row = wm * 32 + tm * 16 + r + ((lane >> 4) << 3);
          sP[row * LP + col] = (bf16)(__expf(sacc[tm][tn][r] * scale - mm) * iz);
        }
      }
    __syncthreads();

    // out += P (128 x 64) @ V (64 x 64)
#pragma unroll
    for (int k0 = 0; k0 < 64; k0 += 32) {
      v16bf bfrag[2];
#pragma unroll
      for (int tn = 0; tn < 2; ++tn)
        bfrag[tn] = frag_b_ldsT(sVT, wn * 32 + tn * 16, LV, k0, lane);
#pragma unroll
      for (int tm = 0; tm < 2; ++tm) {
        v16bf afrag = frag_a_lds(sP, wm * 32 + tm * 16, LP, k0, lane);
#pragma unroll
        for (int tn = 0; tn < 2; ++tn)
          oacc[tm][tn] = wmma_bf16(afrag, bfrag[tn], oacc[tm][tn]);
      }
    }
  }

  // heads[b, q, h, v] (bf16 for the Wo GEMM)
#pragma unroll
  for (int tm = 0; tm < 2; ++tm)
#pragma unroll
    for (int tn = 0; tn < 2; ++tn) {
      const int col = wn * 32 + tn * 16 + (lane & 15);
#pragma unroll
      for (int r = 0; r < 8; ++r) {
        const int row = wm * 32 + tm * 16 + r + ((lane >> 4) << 3);
        const long q = q0 + row;
        heads[(((long)b * T + q) * H + h) * 64 + col] = (bf16)oacc[tm][tn][r];
      }
    }
}

// ---------------------------------------------------------------------------
// LayerNorm over joint (T, DMODEL) per batch: deterministic 2-stage reduction.
// ---------------------------------------------------------------------------
__global__ __launch_bounds__(256) void ln_part_kernel(
    const float* __restrict__ a, const float* __restrict__ b,
    float* __restrict__ part, long perBatch) {
  __shared__ float rs[256], rq[256];
  const long base = (long)blockIdx.y * perBatch;
  float s = 0.f, q = 0.f;
  for (long i = (long)blockIdx.x * blockDim.x + threadIdx.x; i < perBatch;
       i += (long)gridDim.x * blockDim.x) {
    float v = a[base + i] + b[base + i];
    s += v; q += v * v;
  }
  rs[threadIdx.x] = s; rq[threadIdx.x] = q;
  __syncthreads();
  for (int st = 128; st > 0; st >>= 1) {
    if ((int)threadIdx.x < st) {
      rs[threadIdx.x] += rs[threadIdx.x + st];
      rq[threadIdx.x] += rq[threadIdx.x + st];
    }
    __syncthreads();
  }
  if (threadIdx.x == 0) {
    long p = (long)blockIdx.y * gridDim.x + blockIdx.x;
    part[p * 2] = rs[0]; part[p * 2 + 1] = rq[0];
  }
}

__global__ void ln_finalize_kernel(const float* __restrict__ part,
                                   float* __restrict__ stats, int nparts) {
  __shared__ float rs[64], rq[64];
  const int t = threadIdx.x;
  float s = 0.f, q = 0.f;
  for (int i = t; i < nparts; i += 64) {
    s += part[((long)blockIdx.x * nparts + i) * 2];
    q += part[((long)blockIdx.x * nparts + i) * 2 + 1];
  }
  rs[t] = s; rq[t] = q;
  __syncthreads();
  for (int st = 32; st > 0; st >>= 1) {
    if (t < st) { rs[t] += rs[t + st]; rq[t] += rq[t + st]; }
    __syncthreads();
  }
  if (t == 0) { stats[blockIdx.x * 2] = rs[0]; stats[blockIdx.x * 2 + 1] = rq[0]; }
}

__global__ __launch_bounds__(256) void ln_apply_kernel(
    const float* __restrict__ a, const float* __restrict__ b,
    const float* __restrict__ stats, float* __restrict__ outf,
    bf16* __restrict__ outbf, long perBatch) {
  const long base = (long)blockIdx.y * perBatch;
  const float inv_n = 1.f / (float)perBatch;
  const float mean = stats[blockIdx.y * 2] * inv_n;
  const float var = stats[blockIdx.y * 2 + 1] * inv_n - mean * mean;
  const float inv = rsqrtf(var + 1e-5f);
  for (long i = (long)blockIdx.x * blockDim.x + threadIdx.x; i < perBatch;
       i += (long)gridDim.x * blockDim.x) {
    float o = (a[base + i] + b[base + i] - mean) * inv;
    if (outf)  outf[base + i] = o;
    if (outbf) outbf[base + i] = (bf16)o;
  }
}

// ---------------------------------------------------------------------------
// Host launcher
// ---------------------------------------------------------------------------
extern "C" void kernel_launch(void* const* d_in, const int* in_sizes, int n_in,
                              void* d_out, int out_size, void* d_ws, size_t ws_size,
                              hipStream_t stream) {
  (void)in_sizes; (void)n_in; (void)out_size; (void)ws_size;
  const int B = 2, T = 2048, D = 1024, H = 16, DK = 64, FF = 4096;

  const float* x  = (const float*)d_in[0];
  const float* Wq = (const float*)d_in[1];
  const float* Wk = (const float*)d_in[2];
  const float* Wv = (const float*)d_in[3];
  const float* Wo = (const float*)d_in[4];
  const float* W1 = (const float*)d_in[5];
  const float* b1 = (const float*)d_in[6];
  const float* W2 = (const float*)d_in[7];
  const float* b2 = (const float*)d_in[8];
  float* out = (float*)d_out;

  char* ws = (char*)d_ws;
  size_t off = 0;
  auto alloc = [&](size_t bytes) -> char* {
    char* p = ws + off;
    off = (off + bytes + 255) & ~(size_t)255;
    return p;
  };
  bf16*  xbf    = (bf16*)alloc((size_t)B * T * D * 2);
  bf16*  wqb    = (bf16*)alloc((size_t)H * D * DK * 2);
  bf16*  wkb    = (bf16*)alloc((size_t)H * D * DK * 2);
  bf16*  wvb    = (bf16*)alloc((size_t)H * D * DK * 2);
  bf16*  wob    = (bf16*)alloc((size_t)D * D * 2);
  bf16*  w1b    = (bf16*)alloc((size_t)D * FF * 2);
  bf16*  w2b    = (bf16*)alloc((size_t)FF * D * 2);
  bf16*  qb     = (bf16*)alloc((size_t)B * H * T * DK * 2);
  bf16*  kb     = (bf16*)alloc((size_t)B * H * T * DK * 2);
  bf16*  vb     = (bf16*)alloc((size_t)B * H * T * DK * 2);
  float* statsA = (float*)alloc((size_t)B * H * T * 2 * 4);
  bf16*  headsb = (bf16*)alloc((size_t)B * T * D * 2);
  float* attout = (float*)alloc((size_t)B * T * D * 4);
  float* o1f    = (float*)alloc((size_t)B * T * D * 4);
  bf16*  o1b    = (bf16*)alloc((size_t)B * T * D * 2);
  bf16*  ffh    = (bf16*)alloc((size_t)B * T * FF * 2);
  float* ff2    = (float*)alloc((size_t)B * T * D * 4);
  float* lnpart = (float*)alloc((size_t)B * 64 * 2 * 4);
  float* lnst   = (float*)alloc(64 * 4);

  const dim3 blk(256);

  // f32 -> bf16 conversions
  f32_to_bf16_kernel<<<1024, blk, 0, stream>>>(x,  xbf, (long)B * T * D);
  f32_to_bf16_kernel<<<512,  blk, 0, stream>>>(Wq, wqb, (long)H * D * DK);
  f32_to_bf16_kernel<<<512,  blk, 0, stream>>>(Wk, wkb, (long)H * D * DK);
  f32_to_bf16_kernel<<<512,  blk, 0, stream>>>(Wv, wvb, (long)H * D * DK);
  f32_to_bf16_kernel<<<512,  blk, 0, stream>>>(Wo, wob, (long)D * D);
  f32_to_bf16_kernel<<<1024, blk, 0, stream>>>(W1, w1b, (long)D * FF);
  f32_to_bf16_kernel<<<1024, blk, 0, stream>>>(W2, w2b, (long)FF * D);

  // QKV projections: per (b,h): x_b (T x D) @ W*[h] (D x 64) -> (bh, T, 64)
  {
    dim3 g(T / 128, 1, B * H);
    gemm_bf16_kernel<0><<<g, blk, 0, stream>>>(xbf, wqb, qb, nullptr, T, DK, D,
        (long)T * D, H, (long)D * DK, H, (long)T * DK);
    gemm_bf16_kernel<0><<<g, blk, 0, stream>>>(xbf, wkb, kb, nullptr, T, DK, D,
        (long)T * D, H, (long)D * DK, H, (long)T * DK);
    gemm_bf16_kernel<0><<<g, blk, 0, stream>>>(xbf, wvb, vb, nullptr, T, DK, D,
        (long)T * D, H, (long)D * DK, H, (long)T * DK);
  }

  // Attention (softmax over query axis): stats pass + fused AV pass
  attn_stats_kernel<<<dim3(T / 64, B * H), blk, 0, stream>>>(qb, kb, statsA, T);
  attn_av_kernel<<<dim3(T / 128, B * H), blk, 0, stream>>>(qb, kb, vb, statsA,
                                                           headsb, T, H);

  // att_out = heads @ Wo  (B*T x D, f32 out)
  gemm_bf16_kernel<1><<<dim3(B * T / 128, D / 64, 1), blk, 0, stream>>>(
      headsb, wob, attout, nullptr, B * T, D, D, 0L, 1, 0L, 1, 0L);

  // o1 = LayerNorm_{(T,D)}(x + att_out)
  ln_part_kernel<<<dim3(64, B), blk, 0, stream>>>(x, attout, lnpart, (long)T * D);
  ln_finalize_kernel<<<B, 64, 0, stream>>>(lnpart, lnst, 64);
  ln_apply_kernel<<<dim3(64, B), blk, 0, stream>>>(x, attout, lnst, o1f, o1b,
                                                   (long)T * D);

  // ff = relu(o1 @ W1 + b1) @ W2 + b2
  gemm_bf16_kernel<2><<<dim3(B * T / 128, FF / 64, 1), blk, 0, stream>>>(
      o1b, w1b, ffh, b1, B * T, FF, D, 0L, 1, 0L, 1, 0L);
  gemm_bf16_kernel<3><<<dim3(B * T / 128, D / 64, 1), blk, 0, stream>>>(
      ffh, w2b, ff2, b2, B * T, D, FF, 0L, 1, 0L, 1, 0L);

  // out = LayerNorm_{(T,D)}(o1 + ff)
  ln_part_kernel<<<dim3(64, B), blk, 0, stream>>>(o1f, ff2, lnpart, (long)T * D);
  ln_finalize_kernel<<<B, 64, 0, stream>>>(lnpart, lnst + 8, 64);
  ln_apply_kernel<<<dim3(64, B), blk, 0, stream>>>(o1f, ff2, lnst + 8, out,
                                                   nullptr, (long)T * D);
}
